// V6LayerConfigurable_39900246180553
// MI455X (gfx1250) — matrix operand order repaired and checked by reading, not compiled, not run on hardware
//
#include <hip/hip_runtime.h>
#include <hip/hip_bf16.h>
#include <math.h>

// Problem constants (match reference)
#define Bn 4
#define Tn 2048
#define Cn 1024
#define DKn 128
#define Nanch 8

// Padded LDS row stride (in bf16 elems): 40*2=80B, 16B-aligned, and
// 20*n mod 64 is distinct for n=0..15 -> conflict-free b128 frag loads.
#define LPAD 40

typedef __attribute__((ext_vector_type(16))) __bf16 v16bf;
typedef __attribute__((ext_vector_type(8)))  float  v8f;

union FragB { v16bf v; uint4 q[2]; };
union FragW { v16bf v; __bf16 e[16]; };

__device__ __forceinline__ float sigm(float x) { return 1.0f / (1.0f + __expf(-x)); }

__device__ __forceinline__ void wait_async0() {
  asm volatile("s_wait_asynccnt 0" ::: "memory");
}

__device__ __forceinline__ void async_b128_to_lds(unsigned lds_off, const void* gaddr) {
  asm volatile("global_load_async_to_lds_b128 %0, %1, off"
               :: "v"(lds_off), "v"((unsigned long long)(size_t)gaddr)
               : "memory");
}

// ---------------------------------------------------------------------------
// WMMA GEMM: out(M,N) = EPI(A(M,K) @ W(K,N) + bias)
// A: fp32, split to bf16 hi/lo in-kernel. W: pre-converted transposed bf16
// hi/lo (Wt[n][k]), staged via global_load_async_to_lds_b128 (ASYNCcnt).
// 3 WMMA products (hi*hi + hi*lo + lo*hi) for ~fp32 accuracy.
// Block = 256 threads (8 waves). Tile 128(M) x 64(N), K step 32,
// double-buffered LDS, one barrier per K step.
// EPI: 0 = none, 1 = tanh, 2 = sigmoid, 3 = blend: out = p + bl*(sigmoid(r)-p)
// ---------------------------------------------------------------------------
template <int EPI>
__global__ __launch_bounds__(256) void gemm_wmma(
    const float* __restrict__ A,
    const __bf16* __restrict__ Wthi, const __bf16* __restrict__ Wtlo,
    const float* __restrict__ bias, float* __restrict__ out,
    const float* __restrict__ blendRaw, int M, int N, int K)
{
  __shared__ __bf16 Ah[2][128][LPAD];
  __shared__ __bf16 Al[2][128][LPAD];
  __shared__ __bf16 Bh[2][64][LPAD];
  __shared__ __bf16 Bl[2][64][LPAD];

  const int tid = threadIdx.x;
  const int l   = tid & 31;
  const int w   = tid >> 5;
  const int m0  = blockIdx.y * 128;
  const int n0  = blockIdx.x * 64;

  const int ak = (tid & 7) * 4;       // k offset for A staging
  const int am = tid >> 3;            // row group for A staging (0..31)
  const int br = (tid >> 2) & 63;     // B row (n) for async staging
  const int bc = (tid & 3) * 8;       // B k-offset (bf16), 16B chunk

  float4 areg[4];

  auto loadA = [&](int k0) {
#pragma unroll
    for (int p = 0; p < 4; p++)
      areg[p] = *(const float4*)(A + (size_t)(m0 + am + p * 32) * K + k0 + ak);
  };
  auto storeA = [&](int buf) {
#pragma unroll
    for (int p = 0; p < 4; p++) {
      int row = am + p * 32;
      float f[4] = {areg[p].x, areg[p].y, areg[p].z, areg[p].w};
#pragma unroll
      for (int u = 0; u < 4; u++) {
        __bf16 h = (__bf16)f[u];
        Ah[buf][row][ak + u] = h;
        Al[buf][row][ak + u] = (__bf16)(f[u] - (float)h);
      }
    }
  };
  auto stageB = [&](int k0, int buf) {
    const __bf16* ghi = Wthi + (size_t)(n0 + br) * K + k0 + bc;
    const __bf16* glo = Wtlo + (size_t)(n0 + br) * K + k0 + bc;
    async_b128_to_lds((unsigned)(size_t)&Bh[buf][br][bc], ghi);
    async_b128_to_lds((unsigned)(size_t)&Bl[buf][br][bc], glo);
  };

  v8f acc[4] = {};

  // Prologue: stage tile 0 into buffer 0
  loadA(0);
  stageB(0, 0);
  storeA(0);
  wait_async0();
  __syncthreads();

  const int arow  = w * 16 + (l & 15);
  const int akoff = (l >> 4) * 8;
  const int bkoff = (l >> 4) * 8;

  int cur = 0;
  for (int k0 = 0; k0 < K; k0 += 32) {
    const bool more = (k0 + 32) < K;
    if (more) {
      loadA(k0 + 32);
      stageB(k0 + 32, cur ^ 1);
    }

    // A fragments (16x32): lane=row, e0..7 -> k=koff+e, e8..15 -> k=koff+16+e
    FragB ah, al;
    ah.q[0] = *(const uint4*)&Ah[cur][arow][akoff];
    ah.q[1] = *(const uint4*)&Ah[cur][arow][akoff + 16];
    al.q[0] = *(const uint4*)&Al[cur][arow][akoff];
    al.q[1] = *(const uint4*)&Al[cur][arow][akoff + 16];

#pragma unroll
    for (int nt = 0; nt < 4; nt++) {
      FragB bh, bl;
      const int bcol = nt * 16 + (l & 15);
      bh.q[0] = *(const uint4*)&Bh[cur][bcol][bkoff];
      bh.q[1] = *(const uint4*)&Bh[cur][bcol][bkoff + 16];
      bl.q[0] = *(const uint4*)&Bl[cur][bcol][bkoff];
      bl.q[1] = *(const uint4*)&Bl[cur][bcol][bkoff + 16];

      acc[nt] = __builtin_amdgcn_wmma_f32_16x16x32_bf16(false, ah.v, false, bh.v,
                                                        (short)0, acc[nt], false, false);
      acc[nt] = __builtin_amdgcn_wmma_f32_16x16x32_bf16(false, ah.v, false, bl.v,
                                                        (short)0, acc[nt], false, false);
      acc[nt] = __builtin_amdgcn_wmma_f32_16x16x32_bf16(false, al.v, false, bh.v,
                                                        (short)0, acc[nt], false, false);
    }

    if (more) {
      storeA(cur ^ 1);
      wait_async0();
    }
    __syncthreads();
    cur ^= 1;
  }

  float blv = 0.0f;
  if (EPI == 3) blv = sigm(blendRaw[0]);

  const int rbase = m0 + w * 16 + ((l >> 4) << 3);
#pragma unroll
  for (int nt = 0; nt < 4; nt++) {
    const int col = n0 + nt * 16 + (l & 15);
    const float bv = bias ? bias[col] : 0.0f;
#pragma unroll
    for (int v = 0; v < 8; v++) {
      float r = acc[nt][v] + bv;
      size_t idx = (size_t)(rbase + v) * N + col;
      if (EPI == 1) {
        r = tanhf(r);
      } else if (EPI == 2) {
        r = sigm(r);
      } else if (EPI == 3) {
        float g = sigm(r);
        float p = out[idx];
        r = p + blv * (g - p);
      }
      out[idx] = r;
    }
  }
}

// ---------------------------------------------------------------------------
// Weight pre-pass: transposed bf16 hi/lo split. Whi/Wlo are (N x K);
// Whi[n*K+k] = bf16(W[k*N+n]); Wlo = bf16(residual).
// ---------------------------------------------------------------------------
__global__ void convert_w_t(const float* __restrict__ W,
                            __bf16* __restrict__ Whi, __bf16* __restrict__ Wlo,
                            int K, int N)
{
  int idx = blockIdx.x * 256 + threadIdx.x;  // N*K threads, k fastest
  int n = idx / K, k = idx % K;
  float f = W[(size_t)k * N + n];
  __bf16 h = (__bf16)f;
  Whi[idx] = h;
  Wlo[idx] = (__bf16)(f - (float)h);
}

// ---------------------------------------------------------------------------
// W transpose (for recurrence streaming): Wt[j][i] = W[i][j]
// ---------------------------------------------------------------------------
__global__ void transpose1024(const float* __restrict__ in, float* __restrict__ out)
{
  int idx = blockIdx.x * 256 + threadIdx.x;
  int i = idx / Cn, j = idx % Cn;
  out[(size_t)j * Cn + i] = in[idx];
}

// ---------------------------------------------------------------------------
// Gated EMA recurrence: h_t = alpha*h_{t-1} + sigmoid(gx_t + h_{t-1}@Wwh)*xp_t
// One 1024-thread block per batch; h lives in LDS; Wt is transposed Wwh so
// each lane streams a contiguous 4KB row (L2-resident: 4MB << 192MB).
// ---------------------------------------------------------------------------
__global__ __launch_bounds__(1024) void recurrence(
    const float* __restrict__ gx, const float* __restrict__ xp,
    const float* __restrict__ Wt, float* __restrict__ ctx,
    const float* __restrict__ tauPtr)
{
  __shared__ float h[Cn];
  const int j = threadIdx.x;
  const int b = blockIdx.x;
  const float tau = tauPtr[0];
  const float alpha = sigm(-tau);  // exp(-softplus(tau)) == sigmoid(-tau)

  h[j] = 0.0f;
  __syncthreads();

  const float4* wrow = (const float4*)(Wt + (size_t)j * Cn);
  const float4* hp   = (const float4*)h;

  for (int t = 0; t < Tn; t++) {
    size_t off = ((size_t)b * Tn + t) * Cn + j;
    float gxv = gx[off];
    float xpv = xp[off];
    if (t + 1 < Tn) {
      __builtin_prefetch(gx + off + Cn, 0, 1);
      __builtin_prefetch(xp + off + Cn, 0, 1);
    }
    float s = 0.0f;
#pragma unroll 8
    for (int i = 0; i < Cn / 4; i++) {
      float4 wv = wrow[i];
      float4 hv = hp[i];
      s += wv.x * hv.x + wv.y * hv.y + wv.z * hv.z + wv.w * hv.w;
    }
    float wg = sigm(gxv + s);
    float hn = alpha * h[j] + wg * xpv;
    __syncthreads();
    h[j] = hn;
    ctx[off] = hn;
    __syncthreads();
  }
}

// ---------------------------------------------------------------------------
// Content gate: cg[b,t] = sigmoid(max_{k<=t} (q_t . k_k) / sqrt(DK))
// WMMA q@k^T in 16x16 tiles; per-lane running max; causal mask on diagonal
// tile via the C/D VGPR layout; one cross-lane reduction at the end.
// Block = 256 (8 waves); each wave owns a 16-row query block.
// ---------------------------------------------------------------------------
__global__ __launch_bounds__(256) void cg_rowmax(
    const float* __restrict__ q, const float* __restrict__ k,
    float* __restrict__ cg)
{
  const int l  = threadIdx.x & 31;
  const int w  = threadIdx.x >> 5;
  const int b  = blockIdx.y;
  const int qb = blockIdx.x * 8 + w;
  const int qbase = qb * 16;

  const float* qB = q + (size_t)b * Tn * DKn;
  const float* kB = k + (size_t)b * Tn * DKn;

  float pmax[8];
#pragma unroll
  for (int v = 0; v < 8; v++) pmax[v] = -3.0e38f;

  const int arow = qbase + (l & 15);
  const int Nc   = l & 15;

  for (int kb = 0; kb <= qb; kb++) {
    const int kbase = kb * 16;
    v8f acc = {};
#pragma unroll
    for (int kc = 0; kc < DKn; kc += 32) {
      // A fragment from q block (16x32)
      const float* ap = qB + (size_t)arow * DKn + kc + ((l >> 4) * 8);
      float4 a0 = *(const float4*)(ap);
      float4 a1 = *(const float4*)(ap + 4);
      float4 a2 = *(const float4*)(ap + 16);
      float4 a3 = *(const float4*)(ap + 20);
      FragW af;
      af.e[0] = (__bf16)a0.x;  af.e[1] = (__bf16)a0.y;
      af.e[2] = (__bf16)a0.z;  af.e[3] = (__bf16)a0.w;
      af.e[4] = (__bf16)a1.x;  af.e[5] = (__bf16)a1.y;
      af.e[6] = (__bf16)a1.z;  af.e[7] = (__bf16)a1.w;
      af.e[8] = (__bf16)a2.x;  af.e[9] = (__bf16)a2.y;
      af.e[10] = (__bf16)a2.z; af.e[11] = (__bf16)a2.w;
      af.e[12] = (__bf16)a3.x; af.e[13] = (__bf16)a3.y;
      af.e[14] = (__bf16)a3.z; af.e[15] = (__bf16)a3.w;
      // B fragment = K^T (32x16): lane = key row, k contiguous in memory
      const float* bp = kB + (size_t)(kbase + (l & 15)) * DKn + kc + ((l >> 4) * 8);
      float4 b0 = *(const float4*)(bp);
      float4 b1 = *(const float4*)(bp + 4);
      float4 b2 = *(const float4*)(bp + 16);
      float4 b3 = *(const float4*)(bp + 20);
      FragW bfm;
      bfm.e[0] = (__bf16)b0.x;  bfm.e[1] = (__bf16)b0.y;
      bfm.e[2] = (__bf16)b0.z;  bfm.e[3] = (__bf16)b0.w;
      bfm.e[4] = (__bf16)b1.x;  bfm.e[5] = (__bf16)b1.y;
      bfm.e[6] = (__bf16)b1.z;  bfm.e[7] = (__bf16)b1.w;
      bfm.e[8] = (__bf16)b2.x;  bfm.e[9] = (__bf16)b2.y;
      bfm.e[10] = (__bf16)b2.z; bfm.e[11] = (__bf16)b2.w;
      bfm.e[12] = (__bf16)b3.x; bfm.e[13] = (__bf16)b3.y;
      bfm.e[14] = (__bf16)b3.z; bfm.e[15] = (__bf16)b3.w;

      acc = __builtin_amdgcn_wmma_f32_16x16x32_bf16(false, af.v, false, bfm.v,
                                                    (short)0, acc, false, false);
    }
    // per-lane running max with causal mask on the diagonal tile
#pragma unroll
    for (int v = 0; v < 8; v++) {
      int Mr = v + ((l >> 4) << 3);
      float s = acc[v];
      if (kb == qb && Nc > Mr) s = -3.0e38f;
      pmax[v] = fmaxf(pmax[v], s);
    }
  }

  // reduce across the 16-lane N group
#pragma unroll
  for (int v = 0; v < 8; v++) {
    float s = pmax[v];
#pragma unroll
    for (int off = 1; off < 16; off <<= 1)
      s = fmaxf(s, __shfl_xor(s, off, 32));
    pmax[v] = s;
  }

  if ((l & 15) == 0) {
    const float scale = 0.08838834764831845f;  // 1/sqrt(128)
    int rowoff = (l >> 4) << 3;
#pragma unroll
    for (int v = 0; v < 8; v++)
      cg[(size_t)b * Tn + qbase + rowoff + v] = sigm(pmax[v] * scale);
  }
}

// ---------------------------------------------------------------------------
// Anchors: addv[b,c] = sum_n sigmoid(scales[n]) * tanh(x[b,pos_n,:]@Wa[:,c]+ba[c])
// ---------------------------------------------------------------------------
__global__ __launch_bounds__(256) void anchor_add(
    const float* __restrict__ x, const float* __restrict__ Wa,
    const float* __restrict__ ba, const float* __restrict__ scalesRaw,
    float* __restrict__ addv)
{
  int idx = blockIdx.x * 256 + threadIdx.x;  // B*C threads
  int b = idx / Cn, c = idx % Cn;
  float accv = 0.0f;
  for (int n = 0; n < Nanch; n++) {
    int p = n * (Tn / Nanch);
    const float* xr = x + ((size_t)b * Tn + p) * Cn;
    float s = 0.0f;
#pragma unroll 4
    for (int i = 0; i < Cn; i++) s += xr[i] * Wa[(size_t)i * Cn + c];
    accv += sigm(scalesRaw[n]) * tanhf(s + ba[c]);
  }
  addv[idx] = accv;
}

// ---------------------------------------------------------------------------
// ctx = ctx * cg[b,t] + addv[b,c]  (both fast and slow)
// ---------------------------------------------------------------------------
__global__ __launch_bounds__(256) void ctx_fixup(
    float* __restrict__ ctxf, float* __restrict__ ctxs,
    const float* __restrict__ cg, const float* __restrict__ addv)
{
  size_t idx = (size_t)blockIdx.x * 256 + threadIdx.x;
  int c = (int)(idx % Cn);
  size_t bt = idx / Cn;
  int b = (int)(bt / Tn);
  float g  = cg[bt];
  float av = addv[(size_t)b * Cn + c];
  ctxf[idx] = ctxf[idx] * g + av;
  ctxs[idx] = ctxs[idx] * g + av;
}

// ---------------------------------------------------------------------------
// Final: y = soma*gate; LayerNorm over C; one 256-thread block per row.
// ---------------------------------------------------------------------------
__global__ __launch_bounds__(256) void final_ln(
    const float* __restrict__ soma, const float* __restrict__ gate,
    const float* __restrict__ lng, const float* __restrict__ lnb,
    float* __restrict__ out)
{
  __shared__ float red[256];
  __shared__ float red2[256];
  int r = blockIdx.x;
  int t = threadIdx.x;
  const float4* sp = (const float4*)(soma + (size_t)r * Cn);
  const float4* gp = (const float4*)(gate + (size_t)r * Cn);
  float4 sv = sp[t], gv = gp[t];
  float4 y = {sv.x * gv.x, sv.y * gv.y, sv.z * gv.z, sv.w * gv.w};
  float s  = y.x + y.y + y.z + y.w;
  float s2 = y.x * y.x + y.y * y.y + y.z * y.z + y.w * y.w;
  red[t] = s; red2[t] = s2;
  __syncthreads();
  for (int o = 128; o > 0; o >>= 1) {
    if (t < o) { red[t] += red[t + o]; red2[t] += red2[t + o]; }
    __syncthreads();
  }
  float mu  = red[0] * (1.0f / Cn);
  float var = red2[0] * (1.0f / Cn) - mu * mu;
  float inv = rsqrtf(var + 1e-5f);
  int c = t * 4;
  float* op = out + (size_t)r * Cn + c;
  op[0] = (y.x - mu) * inv * lng[c + 0] + lnb[c + 0];
  op[1] = (y.y - mu) * inv * lng[c + 1] + lnb[c + 1];
  op[2] = (y.z - mu) * inv * lng[c + 2] + lnb[c + 2];
  op[3] = (y.w - mu) * inv * lng[c + 3] + lnb[c + 3];
}

// ---------------------------------------------------------------------------
extern "C" void kernel_launch(void* const* d_in, const int* in_sizes, int n_in,
                              void* d_out, int out_size, void* d_ws, size_t ws_size,
                              hipStream_t stream)
{
  const float* x        = (const float*)d_in[0];
  const float* W_proj   = (const float*)d_in[1];
  const float* b_proj   = (const float*)d_in[2];
  const float* tau_f    = (const float*)d_in[3];
  const float* tau_s    = (const float*)d_in[4];
  const float* W_wf     = (const float*)d_in[5];
  const float* b_wf     = (const float*)d_in[6];
  const float* W_ws     = (const float*)d_in[7];
  const float* b_ws     = (const float*)d_in[8];
  const float* W_whf    = (const float*)d_in[9];
  const float* W_whs    = (const float*)d_in[10];
  const float* W_cgq    = (const float*)d_in[11];
  const float* W_cgk    = (const float*)d_in[12];
  const float* W_fast   = (const float*)d_in[13];
  const float* b_fast   = (const float*)d_in[14];
  const float* W_slow   = (const float*)d_in[15];
  const float* b_slow   = (const float*)d_in[16];
  const float* W_soma   = (const float*)d_in[17];
  const float* b_soma   = (const float*)d_in[18];
  const float* W_anchor = (const float*)d_in[19];
  const float* b_anchor = (const float*)d_in[20];
  const float* anch     = (const float*)d_in[21];
  const float* blendRaw = (const float*)d_in[22];
  const float* ln_g     = (const float*)d_in[23];
  const float* ln_b     = (const float*)d_in[24];

  float* ws = (float*)d_ws;
  const size_t BTC = (size_t)Bn * Tn * Cn;
  const size_t CC  = (size_t)Cn * Cn;
  const size_t CD  = (size_t)Cn * DKn;

  float* xproj = ws; ws += BTC;
  float* gxf   = ws; ws += BTC;
  float* gxs   = ws; ws += BTC;
  float* ctxf  = ws; ws += BTC;
  float* ctxs  = ws; ws += BTC;
  float* qb    = ws; ws += (size_t)Bn * Tn * DKn;
  float* kb    = ws; ws += (size_t)Bn * Tn * DKn;
  float* Wtf   = ws; ws += CC;
  float* Wts   = ws; ws += CC;
  float* cgb   = ws; ws += (size_t)Bn * Tn;
  float* addv  = ws; ws += (size_t)Bn * Cn;
  // bf16 transposed hi/lo weights (each CC or CD elements of __bf16)
  __bf16* wb = (__bf16*)ws;
  __bf16* WprojH = wb; wb += CC;  __bf16* WprojL = wb; wb += CC;
  __bf16* WwfH   = wb; wb += CC;  __bf16* WwfL   = wb; wb += CC;
  __bf16* WwsH   = wb; wb += CC;  __bf16* WwsL   = wb; wb += CC;
  __bf16* WfastH = wb; wb += CC;  __bf16* WfastL = wb; wb += CC;
  __bf16* WslowH = wb; wb += CC;  __bf16* WslowL = wb; wb += CC;
  __bf16* WsomaH = wb; wb += CC;  __bf16* WsomaL = wb; wb += CC;
  __bf16* WcgqH  = wb; wb += CD;  __bf16* WcgqL  = wb; wb += CD;
  __bf16* WcgkH  = wb; wb += CD;  __bf16* WcgkL  = wb; wb += CD;

  // After the recurrences consume gxf/gxs, reuse them:
  float* soma = gxf;
  float* gate = gxs;

  const int M = Bn * Tn;  // 8192
  const int gcc = (int)(CC / 256);
  const int gcd = (int)(CD / 256);

  // Weight pre-passes
  convert_w_t<<<gcc, 256, 0, stream>>>(W_proj, WprojH, WprojL, Cn, Cn);
  convert_w_t<<<gcc, 256, 0, stream>>>(W_wf,   WwfH,   WwfL,   Cn, Cn);
  convert_w_t<<<gcc, 256, 0, stream>>>(W_ws,   WwsH,   WwsL,   Cn, Cn);
  convert_w_t<<<gcc, 256, 0, stream>>>(W_fast, WfastH, WfastL, Cn, Cn);
  convert_w_t<<<gcc, 256, 0, stream>>>(W_slow, WslowH, WslowL, Cn, Cn);
  convert_w_t<<<gcc, 256, 0, stream>>>(W_soma, WsomaH, WsomaL, Cn, Cn);
  convert_w_t<<<gcd, 256, 0, stream>>>(W_cgq,  WcgqH,  WcgqL,  Cn, DKn);
  convert_w_t<<<gcd, 256, 0, stream>>>(W_cgk,  WcgkH,  WcgkL,  Cn, DKn);
  transpose1024<<<gcc, 256, 0, stream>>>(W_whf, Wtf);
  transpose1024<<<gcc, 256, 0, stream>>>(W_whs, Wts);

  dim3 gC(Cn / 64, M / 128);    // (16, 64)
  dim3 gQ(DKn / 64, M / 128);   // (2, 64)

  gemm_wmma<1><<<gC, 256, 0, stream>>>(x, WprojH, WprojL, b_proj, xproj, nullptr, M, Cn, Cn);
  gemm_wmma<0><<<gC, 256, 0, stream>>>(x, WwfH, WwfL, b_wf, gxf, nullptr, M, Cn, Cn);
  gemm_wmma<0><<<gC, 256, 0, stream>>>(x, WwsH, WwsL, b_ws, gxs, nullptr, M, Cn, Cn);
  gemm_wmma<0><<<gQ, 256, 0, stream>>>(x, WcgqH, WcgqL, nullptr, qb, nullptr, M, DKn, Cn);
  gemm_wmma<0><<<gQ, 256, 0, stream>>>(x, WcgkH, WcgkL, nullptr, kb, nullptr, M, DKn, Cn);

  recurrence<<<Bn, 1024, 0, stream>>>(gxf, xproj, Wtf, ctxf, tau_f);
  recurrence<<<Bn, 1024, 0, stream>>>(gxs, xproj, Wts, ctxs, tau_s);

  dim3 gcg(Tn / 128, Bn);
  cg_rowmax<<<gcg, 256, 0, stream>>>(qb, kb, cgb);

  anchor_add<<<(Bn * Cn) / 256, 256, 0, stream>>>(x, W_anchor, b_anchor, anch, addv);
  ctx_fixup<<<(int)(BTC / 256), 256, 0, stream>>>(ctxf, ctxs, cgb, addv);

  gemm_wmma<2><<<gC, 256, 0, stream>>>(ctxf, WfastH, WfastL, b_fast, gate, nullptr, M, Cn, Cn);
  gemm_wmma<3><<<gC, 256, 0, stream>>>(ctxs, WslowH, WslowL, b_slow, gate, blendRaw, M, Cn, Cn);
  gemm_wmma<1><<<gC, 256, 0, stream>>>(x, WsomaH, WsomaL, b_soma, soma, nullptr, M, Cn, Cn);

  final_ln<<<M, 256, 0, stream>>>(soma, gate, ln_g, ln_b, (float*)d_out);
}
